// ScAgeNet_9431748182649
// MI455X (gfx1250) — matrix operand order repaired and checked by reading, not compiled, not run on hardware
//
#include <hip/hip_runtime.h>
#include <hip/hip_bf16.h>

// ---------------------------------------------------------------------------
// ScAgeNet forward for MI455X (gfx1250, wave32, WMMA + Tensor Data Mover).
// Encoder: 3x [GEMM + eval-BN + ReLU], split-bf16 WMMA (hi*hi + hi*lo + lo*hi).
// Activations stored as bf16 hi/lo planes; A/W tiles staged via TDM
// (tensor_load_to_lds) with LDS padding for bank-conflict-free fragments.
// Pipelining: layer1 = register-pipelined A loads; layers2/3 = LDS double-
// buffered A with partial s_wait_tensorcnt. Heads: routed, TDM gather mode.
// ---------------------------------------------------------------------------

typedef __attribute__((ext_vector_type(16))) __bf16        v16bf;
typedef __attribute__((ext_vector_type(8)))  float         v8f;
typedef __attribute__((ext_vector_type(4)))  unsigned int  v4u;
typedef __attribute__((ext_vector_type(8)))  int           v8i;
typedef __attribute__((ext_vector_type(4)))  int           v4i;

#define BCELLS 32768
#define NEXP   29
#define LN_EPS 1e-5f
#define SWB    36   // padded k-stride (bf16 elems): %4==0 (align), %16!=0 (banks)

#define USE_TDM __has_builtin(__builtin_amdgcn_tensor_load_to_lds)

#if USE_TDM
#define TDM_G0_PLAIN   1u
#define TDM_G0_GATHER  (1u | (1u << 31))          // gather, 16-bit indices
#define TDM_G1_B8      (3u << 16)                  // 8B elems, no pad
// bf16 elems, pad_enable, pad every 16 DWORDs (64B = one 32-elem row) by 2 DWORDs
#define TDM_G1_BF16PAD ((1u << 16) | (1u << 20) | (3u << 22) | (1u << 25))

__device__ __forceinline__ int rfl(int x) { return __builtin_amdgcn_readfirstlane(x); }

__device__ __forceinline__ v8i tdm_g1(uint32_t g1d0, uint32_t tile0, uint32_t tile1,
                                      uint32_t stride0) {
    const uint32_t dim0 = 1u << 30, dim1 = 1u << 30;  // "huge" -> no OOB clipping
    v8i g1;
    g1[0] = (int)g1d0;
    g1[1] = (int)((dim0 & 0xffffu) << 16);                  // tensor_dim0 lo
    g1[2] = (int)((dim0 >> 16) | ((dim1 & 0xffffu) << 16)); // dim0 hi | dim1 lo
    g1[3] = (int)((dim1 >> 16) | (tile0 << 16));            // dim1 hi | tile_dim0
    g1[4] = (int)tile1;                                     // tile_dim1 (dim2=0)
    g1[5] = (int)stride0;                                   // tensor_dim0_stride lo
    g1[6] = 0;
    g1[7] = 0;
    return g1;
}

// Hoistable descriptor: invariant groups built once; lds + global patched/issue.
struct Tdm {
    unsigned flags;
    v8i g1;
    v4i g2, g3;
    __device__ __forceinline__ void init(unsigned f, uint32_t g1d0, uint32_t tile0,
                                         uint32_t tile1, uint32_t stride0) {
        flags = f;
        g1 = tdm_g1(g1d0, tile0, tile1, stride0);
        g2 = (v4i){};
        g3 = (v4i){};
    }
    __device__ __forceinline__ void issue(void* lds, const void* gsrc) const {
        uint64_t ga = (uint64_t)(uintptr_t)gsrc;
        v4u g0;
        g0[0] = flags;
        g0[1] = (uint32_t)(uintptr_t)lds;
        g0[2] = (uint32_t)ga;
        g0[3] = (uint32_t)(ga >> 32) | (2u << 30);   // addr hi | type=2
        v8i z8 = {};
        __builtin_amdgcn_tensor_load_to_lds(g0, g1, g2, g3, z8, 0);
    }
};

// one-shot build+issue (head kernel)
__device__ __forceinline__ void tdm_load(uint32_t g0d0, void* lds, const void* gsrc,
                                         uint32_t g1d0, uint32_t tile0, uint32_t tile1,
                                         uint32_t stride0, v4i g2, v4i g3) {
    uint64_t ga = (uint64_t)(uintptr_t)gsrc;
    v4u g0;
    g0[0] = g0d0;
    g0[1] = (uint32_t)(uintptr_t)lds;
    g0[2] = (uint32_t)ga;
    g0[3] = (uint32_t)(ga >> 32) | (2u << 30);
    v8i g1 = tdm_g1(g1d0, tile0, tile1, stride0);
    v8i z8 = {};
    __builtin_amdgcn_tensor_load_to_lds(g0, g1, g2, g3, z8, 0);
}
#endif

// ---------------- fallback: per-lane async copies ----------------------------
__device__ __forceinline__ void copy_tile(void* lds_dst, const void* gsrc,
                                          int bytes, int tid) {
    uint32_t l = (uint32_t)(uintptr_t)lds_dst;
    uint64_t g = (uint64_t)(uintptr_t)gsrc;
    for (int off = tid * 16; off < bytes; off += 256 * 16) {
        asm volatile("global_load_async_to_lds_b128 %0, %1, off"
                     :: "v"(l + off), "v"(g + off)
                     : "memory");
    }
}

__device__ __forceinline__ void stage_wait0() {
#if USE_TDM
#if __has_builtin(__builtin_amdgcn_s_wait_tensorcnt)
    __builtin_amdgcn_s_wait_tensorcnt(0);
#else
    asm volatile("s_wait_tensorcnt 0" ::: "memory");
#endif
#else
#if __has_builtin(__builtin_amdgcn_s_wait_asynccnt)
    __builtin_amdgcn_s_wait_asynccnt(0);
#else
    asm volatile("s_wait_asynccnt 0" ::: "memory");
#endif
#endif
}

#if USE_TDM
__device__ __forceinline__ void tdm_wait2() {
#if __has_builtin(__builtin_amdgcn_s_wait_tensorcnt)
    __builtin_amdgcn_s_wait_tensorcnt(2);
#else
    asm volatile("s_wait_tensorcnt 2" ::: "memory");
#endif
}
#endif

// ---------------- non-hoisted staging (head kernel + fallbacks) --------------
__device__ __forceinline__ void stage_contig(void* lds, const void* g, int bytes,
                                             int tid, int wave) {
#if USE_TDM
    if (wave == 0) {
        v4i z = {};
        tdm_load(TDM_G0_PLAIN, lds, g, TDM_G1_B8, (uint32_t)bytes >> 3, 0,
                 (uint32_t)bytes >> 3, z, z);
    }
#else
    copy_tile(lds, g, bytes, tid);
#endif
}

__device__ __forceinline__ void stage_tile_bf_sync(void* lds, const __bf16* g,
                                                   int rows, int gstride, int tid) {
    for (int i = tid; i < rows * 8; i += 256) {
        int r = i >> 3, c = (i & 7) * 4;                // 4-bf16 (8B) chunks
        *(uint2*)((char*)lds + r * (SWB * 2) + c * 2) =
            *(const uint2*)(g + (size_t)r * gstride + c);
    }
}

// gather 64 rows (4x16 indices) of 32 bf16 from g (already offset by k0)
__device__ __forceinline__ void stage_gather_bf(void* lds, const __bf16* g,
                                                const int* rowIdx, int tid, int wave,
                                                const v4i* gi2, const v4i* gi3) {
#if USE_TDM
    if (wave == 0) {
#pragma unroll
        for (int d = 0; d < 4; ++d)
            tdm_load(TDM_G0_GATHER, (char*)lds + d * 16 * SWB * 2, g,
                     TDM_G1_BF16PAD, 32, 16, 256, gi2[d], gi3[d]);
    }
#else
    for (int i = tid; i < 64 * 8; i += 256) {
        int r = i >> 3, c = (i & 7) * 4;
        int cell = rowIdx[r]; if (cell < 0) cell = 0;
        *(uint2*)((char*)lds + r * (SWB * 2) + c * 2) =
            *(const uint2*)(g + (size_t)cell * 256 + c);
    }
#endif
}

// ---------------- WMMA helpers ----------------------------------------------
__device__ __forceinline__ v16bf ldfrag_bf(const __bf16* p0, const __bf16* p1) {
    union { v16bf v; uint2 u[4]; } t;
    t.u[0] = *(const uint2*)(p0);
    t.u[1] = *(const uint2*)(p0 + 4);
    t.u[2] = *(const uint2*)(p1);
    t.u[3] = *(const uint2*)(p1 + 4);
    return t.v;
}

__device__ __forceinline__ v8f wmma3(v16bf ah, v16bf al, v16bf bh, v16bf bl, v8f c) {
    c = __builtin_amdgcn_wmma_f32_16x16x32_bf16(false, ah, false, bh, (short)0, c, false, false);
    c = __builtin_amdgcn_wmma_f32_16x16x32_bf16(false, ah, false, bl, (short)0, c, false, false);
    c = __builtin_amdgcn_wmma_f32_16x16x32_bf16(false, al, false, bh, (short)0, c, false, false);
    return c;
}

__device__ __forceinline__ void cvt4(float v0, float v1, float v2, float v3,
                                     uint2& hi, uint2& lo) {
    union { __bf16 b[4]; uint2 u; } ph, pl;
    ph.b[0] = (__bf16)v0; pl.b[0] = (__bf16)(v0 - (float)ph.b[0]);
    ph.b[1] = (__bf16)v1; pl.b[1] = (__bf16)(v1 - (float)ph.b[1]);
    ph.b[2] = (__bf16)v2; pl.b[2] = (__bf16)(v2 - (float)ph.b[2]);
    ph.b[3] = (__bf16)v3; pl.b[3] = (__bf16)(v3 - (float)ph.b[3]);
    hi = ph.u; lo = pl.u;
}

// ---------------------------------------------------------------------------
// Weight pre-pack: f32 -> per (e, cb, ks) contiguous [BN][SWB] hi + [BN][SWB] lo
// ---------------------------------------------------------------------------
__global__ void pack_w_kernel(const float* __restrict__ W, __bf16* __restrict__ out,
                              int K, int N, int BN, int E) {
    const int nks = (K + 31) >> 5;
    const int ncb = N / BN;
    const size_t total = (size_t)ncb * nks * BN * SWB * E;
    const size_t blkSz = (size_t)2 * BN * SWB;
    for (size_t i = (size_t)blockIdx.x * blockDim.x + threadIdx.x; i < total;
         i += (size_t)gridDim.x * blockDim.x) {
        int kk = (int)(i % SWB);
        size_t t = i / SWB;
        int col = (int)(t % BN); t /= BN;
        int ks  = (int)(t % nks); t /= nks;
        int cb  = (int)(t % ncb);
        int e   = (int)(t / ncb);
        float v = 0.f;
        int gk = ks * 32 + kk;
        if (kk < 32 && gk < K) v = W[(size_t)e * K * N + (size_t)gk * N + cb * BN + col];
        __bf16 h = (__bf16)v;
        __bf16 l = (__bf16)(v - (float)h);
        __bf16* ob = out + (((size_t)e * ncb + cb) * nks + ks) * blkSz;
        ob[(size_t)col * SWB + kk] = h;
        ob[(size_t)BN * SWB + (size_t)col * SWB + kk] = l;
    }
}

// ---------------------------------------------------------------------------
// Layer 1: A=f32 (x). Register-pipelined A staging; W via hoisted TDM desc.
// ---------------------------------------------------------------------------
template <int K, int N>
__global__ __launch_bounds__(256) void gemm_f32in_kernel(
    const float* __restrict__ A, const __bf16* __restrict__ Wp,
    const float* __restrict__ bias, const float* __restrict__ gamma,
    const float* __restrict__ beta, const float* __restrict__ mean,
    const float* __restrict__ var,
    __bf16* __restrict__ Ohi, __bf16* __restrict__ Olo) {
    constexpr int NKS = (K + 31) / 32;
    __shared__ __align__(16) __bf16 ldsA[2 * 128 * SWB];
    __shared__ __align__(16) __bf16 ldsW[2 * 128 * SWB];

    const int tid  = threadIdx.x;
    const int lane = tid & 31;
    const int wave = tid >> 5;
    const int wm   = wave & 3;
    const int wn   = wave >> 2;
    const int half = lane >> 4;
    const int l16  = lane & 15;
    const int bm0  = blockIdx.y * 128;
    const int bn0  = blockIdx.x * 128;

    v8f acc[2][4] = {};
    float areg[16];

    auto load_a = [&](int ks) {
        const int k0 = ks * 32;
#pragma unroll
        for (int i = 0; i < 4; ++i) {
            int idx = tid + i * 256;
            int r   = idx >> 3;
            int c4  = (idx & 7) * 4;
            const size_t base = (size_t)(bm0 + r) * K;
            float v0 = 0.f, v1 = 0.f, v2 = 0.f, v3 = 0.f;
            if ((K & 3) == 0 && (K % 32 == 0 || k0 + c4 + 3 < K)) {
                float4 t = *(const float4*)(A + base + k0 + c4);
                v0 = t.x; v1 = t.y; v2 = t.z; v3 = t.w;
            } else {
                if (k0 + c4 + 0 < K) v0 = A[base + k0 + c4 + 0];
                if (k0 + c4 + 1 < K) v1 = A[base + k0 + c4 + 1];
                if (k0 + c4 + 2 < K) v2 = A[base + k0 + c4 + 2];
                if (k0 + c4 + 3 < K) v3 = A[base + k0 + c4 + 3];
            }
            areg[i * 4 + 0] = v0; areg[i * 4 + 1] = v1;
            areg[i * 4 + 2] = v2; areg[i * 4 + 3] = v3;
        }
    };
    load_a(0);

#if USE_TDM
    Tdm dW;
    dW.init(TDM_G0_PLAIN, TDM_G1_B8, (2 * 128 * SWB * 2) >> 3, 0,
            (2 * 128 * SWB * 2) >> 3);
#endif

    for (int ks = 0; ks < NKS; ++ks) {
        const __bf16* wsrc = Wp + ((size_t)blockIdx.x * NKS + ks) * (2 * 128 * SWB);
#if USE_TDM
        if (wave == 0) dW.issue(ldsW, wsrc);
#else
        copy_tile(ldsW, wsrc, 2 * 128 * SWB * 2, tid);
#endif
        // split+store A(ks) from regs
#pragma unroll
        for (int i = 0; i < 4; ++i) {
            int idx = tid + i * 256;
            int r   = idx >> 3;
            int c4  = (idx & 7) * 4;
            uint2 h, l;
            cvt4(areg[i * 4], areg[i * 4 + 1], areg[i * 4 + 2], areg[i * 4 + 3], h, l);
            *(uint2*)&ldsA[r * SWB + c4]             = h;
            *(uint2*)&ldsA[128 * SWB + r * SWB + c4] = l;
        }
        stage_wait0();
        __syncthreads();
        if (ks + 1 < NKS) load_a(ks + 1);   // global latency hides behind WMMA

        v16bf ah[2], al[2], bh[4], bl[4];
#pragma unroll
        for (int mt = 0; mt < 2; ++mt) {
            const __bf16* rp = &ldsA[(wm * 32 + mt * 16 + l16) * SWB];
            ah[mt] = ldfrag_bf(rp + 16 * half, rp + 16 * half + 8);
            al[mt] = ldfrag_bf(rp + 128 * SWB + 16 * half, rp + 128 * SWB + 16 * half + 8);
        }
#pragma unroll
        for (int nt = 0; nt < 4; ++nt) {
            const __bf16* cp = &ldsW[(wn * 64 + nt * 16 + l16) * SWB + 16 * half];
            bh[nt] = ldfrag_bf(cp, cp + 8);
            bl[nt] = ldfrag_bf(cp + 128 * SWB, cp + 128 * SWB + 8);
        }
#pragma unroll
        for (int mt = 0; mt < 2; ++mt)
#pragma unroll
            for (int nt = 0; nt < 4; ++nt)
                acc[mt][nt] = wmma3(ah[mt], al[mt], bh[nt], bl[nt], acc[mt][nt]);
        __syncthreads();
    }

#pragma unroll
    for (int nt = 0; nt < 4; ++nt) {
        int n = bn0 + wn * 64 + nt * 16 + l16;
        float sc = gamma[n] / sqrtf(var[n] + LN_EPS);
        float sh = beta[n] + (bias[n] - mean[n]) * sc;
#pragma unroll
        for (int mt = 0; mt < 2; ++mt) {
            int rbase = bm0 + wm * 32 + mt * 16 + 8 * half;
#pragma unroll
            for (int r = 0; r < 8; ++r) {
                float vv = fmaxf(acc[mt][nt][r] * sc + sh, 0.f);
                __bf16 h = (__bf16)vv;
                Ohi[(size_t)(rbase + r) * N + n] = h;
                Olo[(size_t)(rbase + r) * N + n] = (__bf16)(vv - (float)h);
            }
        }
    }
}

// ---------------------------------------------------------------------------
// Layers 2/3: bf16-plane A, LDS double-buffered, pipelined TDM (partial wait).
// ---------------------------------------------------------------------------
template <int K, int N>
__global__ __launch_bounds__(256) void gemm_bf_kernel(
    const __bf16* __restrict__ Ahi, const __bf16* __restrict__ Alo,
    const __bf16* __restrict__ Wp,
    const float* __restrict__ bias, const float* __restrict__ gamma,
    const float* __restrict__ beta, const float* __restrict__ mean,
    const float* __restrict__ var,
    __bf16* __restrict__ Ohi, __bf16* __restrict__ Olo) {
    constexpr int NKS = K / 32;
    __shared__ __align__(16) __bf16 ldsA[2][2 * 128 * SWB];   // double buffer
    __shared__ __align__(16) __bf16 ldsW[2 * 128 * SWB];

    const int tid  = threadIdx.x;
    const int lane = tid & 31;
    const int wave = tid >> 5;
    const int wm   = wave & 3;
    const int wn   = wave >> 2;
    const int half = lane >> 4;
    const int l16  = lane & 15;
    const int bm0  = blockIdx.y * 128;
    const int bn0  = blockIdx.x * 128;

    v8f acc[2][4] = {};

#if USE_TDM
    Tdm dW, dA;
    dW.init(TDM_G0_PLAIN, TDM_G1_B8, (2 * 128 * SWB * 2) >> 3, 0,
            (2 * 128 * SWB * 2) >> 3);
    dA.init(TDM_G0_PLAIN, TDM_G1_BF16PAD, 32, 128, (uint32_t)K);
    if (wave == 0) {                       // prologue: A(0) in flight
        dA.issue(&ldsA[0][0],         Ahi + (size_t)bm0 * K);
        dA.issue(&ldsA[0][128 * SWB], Alo + (size_t)bm0 * K);
    }
#endif

    for (int ks = 0; ks < NKS; ++ks) {
        const int buf = ks & 1;
        const __bf16* wsrc = Wp + ((size_t)blockIdx.x * NKS + ks) * (2 * 128 * SWB);
#if USE_TDM
        if (wave == 0) {
            dW.issue(ldsW, wsrc);
            if (ks + 1 < NKS) {            // prefetch A(ks+1) into other buffer
                const size_t an = (size_t)bm0 * K + (ks + 1) * 32;
                dA.issue(&ldsA[buf ^ 1][0],         Ahi + an);
                dA.issue(&ldsA[buf ^ 1][128 * SWB], Alo + an);
            }
        }
        if (ks + 1 < NKS) tdm_wait2();     // leave the 2 A(ks+1) copies in flight
        else              stage_wait0();
#else
        copy_tile(ldsW, wsrc, 2 * 128 * SWB * 2, tid);
        {
            const size_t a0 = (size_t)bm0 * K + ks * 32;
            stage_tile_bf_sync(&ldsA[buf][0],         Ahi + a0, 128, K, tid);
            stage_tile_bf_sync(&ldsA[buf][128 * SWB], Alo + a0, 128, K, tid);
        }
        stage_wait0();
#endif
        __syncthreads();

        v16bf ah[2], al[2], bh[4], bl[4];
#pragma unroll
        for (int mt = 0; mt < 2; ++mt) {
            const __bf16* rp = &ldsA[buf][(wm * 32 + mt * 16 + l16) * SWB];
            ah[mt] = ldfrag_bf(rp + 16 * half, rp + 16 * half + 8);
            al[mt] = ldfrag_bf(rp + 128 * SWB + 16 * half, rp + 128 * SWB + 16 * half + 8);
        }
#pragma unroll
        for (int nt = 0; nt < 4; ++nt) {
            const __bf16* cp = &ldsW[(wn * 64 + nt * 16 + l16) * SWB + 16 * half];
            bh[nt] = ldfrag_bf(cp, cp + 8);
            bl[nt] = ldfrag_bf(cp + 128 * SWB, cp + 128 * SWB + 8);
        }
#pragma unroll
        for (int mt = 0; mt < 2; ++mt)
#pragma unroll
            for (int nt = 0; nt < 4; ++nt)
                acc[mt][nt] = wmma3(ah[mt], al[mt], bh[nt], bl[nt], acc[mt][nt]);
        __syncthreads();
    }

#pragma unroll
    for (int nt = 0; nt < 4; ++nt) {
        int n = bn0 + wn * 64 + nt * 16 + l16;
        float sc = gamma[n] / sqrtf(var[n] + LN_EPS);
        float sh = beta[n] + (bias[n] - mean[n]) * sc;
#pragma unroll
        for (int mt = 0; mt < 2; ++mt) {
            int rbase = bm0 + wm * 32 + mt * 16 + 8 * half;
#pragma unroll
            for (int r = 0; r < 8; ++r) {
                float vv = fmaxf(acc[mt][nt][r] * sc + sh, 0.f);
                __bf16 h = (__bf16)vv;
                Ohi[(size_t)(rbase + r) * N + n] = h;
                Olo[(size_t)(rbase + r) * N + n] = (__bf16)(vv - (float)h);
            }
        }
    }
}

// ---------------------------------------------------------------------------
__global__ void route_kernel(const int* __restrict__ labels, int* __restrict__ counts,
                             int* __restrict__ lists) {
    int b = blockIdx.x * blockDim.x + threadIdx.x;
    if (b < BCELLS) {
        int e = labels[b];
        int idx = atomicAdd(&counts[e], 1);
        lists[e * BCELLS + idx] = b;
    }
}

// ---------------------------------------------------------------------------
// Per-expert head: 64 cells/block, feats rows pulled by TDM gather mode.
// ---------------------------------------------------------------------------
__global__ __launch_bounds__(256) void head_kernel(
    const __bf16* __restrict__ fHi, const __bf16* __restrict__ fLo,
    const int* __restrict__ counts, const int* __restrict__ lists,
    const __bf16* __restrict__ H1p, const float* __restrict__ Hb1,
    const float* __restrict__ LG1, const float* __restrict__ LB1,
    const __bf16* __restrict__ H2p, const float* __restrict__ Hb2,
    const float* __restrict__ LG2, const float* __restrict__ LB2,
    const float* __restrict__ HWo, const float* __restrict__ Hbo,
    float* __restrict__ preds) {
    __shared__ __align__(16) __bf16 ldsA[2 * 64 * SWB];
    __shared__ __align__(16) __bf16 ldsW[2 * 128 * SWB];   // W slice; later raw t2 f32
    __shared__ __align__(16) float  bufTf[8704];           // raw t1; later t1 bf16 hi/lo
    __shared__ int rowIdx[64];

    const int e = blockIdx.y;
    const int cnt = counts[e];
    const int row0 = blockIdx.x * 64;
    if (row0 >= cnt) return;
    const int nrows = min(64, cnt - row0);

    const int tid  = threadIdx.x;
    const int lane = tid & 31;
    const int wave = tid >> 5;
    const int wm   = wave & 3;
    const int wn   = wave >> 2;
    const int half = lane >> 4;
    const int l16  = lane & 15;

    if (tid < 64) rowIdx[tid] = (tid < nrows) ? lists[e * BCELLS + row0 + tid] : -1;
    __syncthreads();

    // build gather index groups (16-bit packed); uniform-ized via readfirstlane
    v4i gi2[4] = {}, gi3[4] = {};
    if (wave == 0) {
#pragma unroll
        for (int d = 0; d < 4; ++d) {
            int id[16];
#pragma unroll
            for (int j = 0; j < 16; ++j) {
                int c = rowIdx[d * 16 + j];
                id[j] = c < 0 ? 0 : c;           // dup row 0 for pad rows
            }
#pragma unroll
            for (int j = 0; j < 4; ++j) {
#if USE_TDM
                gi2[d][j] = rfl((id[2 * j] & 0xffff) | (id[2 * j + 1] << 16));
                gi3[d][j] = rfl((id[8 + 2 * j] & 0xffff) | (id[8 + 2 * j + 1] << 16));
#else
                gi2[d][j] = (id[2 * j] & 0xffff) | (id[2 * j + 1] << 16);
                gi3[d][j] = (id[8 + 2 * j] & 0xffff) | (id[8 + 2 * j + 1] << 16);
#endif
            }
        }
    }

    // ---- GEMM1: [64,256] @ HW1[e][256,128] ----
    v8f acc1[4] = {};
    for (int ks = 0; ks < 8; ++ks) {
        const int k0 = ks * 32;
        stage_contig(ldsW, H1p + ((size_t)e * 8 + ks) * (2 * 128 * SWB),
                     2 * 128 * SWB * 2, tid, wave);
        stage_gather_bf(ldsA,            fHi + k0, rowIdx, tid, wave, gi2, gi3);
        stage_gather_bf(ldsA + 64 * SWB, fLo + k0, rowIdx, tid, wave, gi2, gi3);
        stage_wait0();
        __syncthreads();

        v16bf ah, al, bh[4], bl[4];
        {
            const __bf16* rp = &ldsA[(wm * 16 + l16) * SWB];
            ah = ldfrag_bf(rp + 16 * half, rp + 16 * half + 8);
            al = ldfrag_bf(rp + 64 * SWB + 16 * half, rp + 64 * SWB + 16 * half + 8);
        }
#pragma unroll
        for (int nt = 0; nt < 4; ++nt) {
            const __bf16* cp = &ldsW[(wn * 64 + nt * 16 + l16) * SWB + 16 * half];
            bh[nt] = ldfrag_bf(cp, cp + 8);
            bl[nt] = ldfrag_bf(cp + 128 * SWB, cp + 128 * SWB + 8);
        }
#pragma unroll
        for (int nt = 0; nt < 4; ++nt) acc1[nt] = wmma3(ah, al, bh[nt], bl[nt], acc1[nt]);
        __syncthreads();
    }
    // raw t1 (with bias) -> bufTf, stride 129
#pragma unroll
    for (int nt = 0; nt < 4; ++nt) {
        int n = wn * 64 + nt * 16 + l16;
        float b1 = Hb1[e * 128 + n];
#pragma unroll
        for (int r = 0; r < 8; ++r) {
            int row = wm * 16 + 8 * half + r;
            bufTf[row * 129 + n] = acc1[nt][r] + b1;
        }
    }
    __syncthreads();

    // ---- LayerNorm(128)+ReLU, two-phase, write bf16 hi/lo in place ----
    __bf16* Thi = (__bf16*)bufTf;     // [64][136]
    __bf16* Tlo = Thi + 64 * 136;
    {
        float g0 = LG1[e * 128 + lane],      g1 = LG1[e * 128 + lane + 32];
        float g2 = LG1[e * 128 + lane + 64], g3 = LG1[e * 128 + lane + 96];
        float c0 = LB1[e * 128 + lane],      c1 = LB1[e * 128 + lane + 32];
        float c2 = LB1[e * 128 + lane + 64], c3 = LB1[e * 128 + lane + 96];
        float xs[8][4], invs[8], mus[8];
#pragma unroll
        for (int rr = 0; rr < 8; ++rr) {
            int row = wave * 8 + rr;
            const float* tp = &bufTf[row * 129];
            float x0 = tp[lane], x1 = tp[lane + 32], x2 = tp[lane + 64], x3 = tp[lane + 96];
            float s = x0 + x1 + x2 + x3;
            float q = x0 * x0 + x1 * x1 + x2 * x2 + x3 * x3;
#pragma unroll
            for (int off = 16; off > 0; off >>= 1) {
                s += __shfl_xor(s, off, 32);
                q += __shfl_xor(q, off, 32);
            }
            float mu = s * (1.f / 128.f);
            float vr = q * (1.f / 128.f) - mu * mu;
            xs[rr][0] = x0; xs[rr][1] = x1; xs[rr][2] = x2; xs[rr][3] = x3;
            mus[rr] = mu; invs[rr] = 1.f / sqrtf(vr + LN_EPS);
        }
        __syncthreads();
#pragma unroll
        for (int rr = 0; rr < 8; ++rr) {
            int row = wave * 8 + rr;
            float mu = mus[rr], inv = invs[rr];
            float y0 = fmaxf((xs[rr][0] - mu) * inv * g0 + c0, 0.f);
            float y1 = fmaxf((xs[rr][1] - mu) * inv * g1 + c1, 0.f);
            float y2 = fmaxf((xs[rr][2] - mu) * inv * g2 + c2, 0.f);
            float y3 = fmaxf((xs[rr][3] - mu) * inv * g3 + c3, 0.f);
            __bf16 h0 = (__bf16)y0, h1 = (__bf16)y1, h2 = (__bf16)y2, h3 = (__bf16)y3;
            Thi[row * 136 + lane]      = h0; Tlo[row * 136 + lane]      = (__bf16)(y0 - (float)h0);
            Thi[row * 136 + lane + 32] = h1; Tlo[row * 136 + lane + 32] = (__bf16)(y1 - (float)h1);
            Thi[row * 136 + lane + 64] = h2; Tlo[row * 136 + lane + 64] = (__bf16)(y2 - (float)h2);
            Thi[row * 136 + lane + 96] = h3; Tlo[row * 136 + lane + 96] = (__bf16)(y3 - (float)h3);
        }
    }
    __syncthreads();

    // ---- GEMM2: [64,128] @ HW2[e][128,64] ----
    v8f acc2[2] = {};
    for (int ks = 0; ks < 4; ++ks) {
        const int k0 = ks * 32;
        stage_contig(ldsW, H2p + ((size_t)e * 4 + ks) * (2 * 64 * SWB),
                     2 * 64 * SWB * 2, tid, wave);
        stage_wait0();
        __syncthreads();

        v16bf ah, al, bh[2], bl[2];
        {
            const __bf16* rp = &Thi[(wm * 16 + l16) * 136 + k0];
            const __bf16* rq = &Tlo[(wm * 16 + l16) * 136 + k0];
            ah = ldfrag_bf(rp + 16 * half, rp + 16 * half + 8);
            al = ldfrag_bf(rq + 16 * half, rq + 16 * half + 8);
        }
#pragma unroll
        for (int nt = 0; nt < 2; ++nt) {
            const __bf16* cp = &ldsW[(wn * 32 + nt * 16 + l16) * SWB + 16 * half];
            bh[nt] = ldfrag_bf(cp, cp + 8);
            bl[nt] = ldfrag_bf(cp + 64 * SWB, cp + 64 * SWB + 8);
        }
#pragma unroll
        for (int nt = 0; nt < 2; ++nt) acc2[nt] = wmma3(ah, al, bh[nt], bl[nt], acc2[nt]);
        __syncthreads();
    }

    // raw t2 -> reuse ldsW region as f32 [64][68]
    float* t2f = (float*)ldsW;
#pragma unroll
    for (int nt = 0; nt < 2; ++nt) {
        int n = wn * 32 + nt * 16 + l16;
        float b2 = Hb2[e * 64 + n];
#pragma unroll
        for (int r = 0; r < 8; ++r) {
            int row = wm * 16 + 8 * half + r;
            t2f[row * 68 + n] = acc2[nt][r] + b2;
        }
    }
    __syncthreads();

    // ---- LayerNorm(64) + ReLU + dot(HWo) ----
    {
        float g0 = LG2[e * 64 + lane], g1 = LG2[e * 64 + lane + 32];
        float c0 = LB2[e * 64 + lane], c1 = LB2[e * 64 + lane + 32];
        float w0 = HWo[e * 64 + lane], w1 = HWo[e * 64 + lane + 32];
        float bo = Hbo[e];
#pragma unroll
        for (int rr = 0; rr < 8; ++rr) {
            int row = wave * 8 + rr;
            const float* tp = &t2f[row * 68];
            float x0 = tp[lane], x1 = tp[lane + 32];
            float s = x0 + x1, q = x0 * x0 + x1 * x1;
#pragma unroll
            for (int off = 16; off > 0; off >>= 1) {
                s += __shfl_xor(s, off, 32);
                q += __shfl_xor(q, off, 32);
            }
            float mu  = s * (1.f / 64.f);
            float vr  = q * (1.f / 64.f) - mu * mu;
            float inv = 1.f / sqrtf(vr + LN_EPS);
            float y0 = fmaxf((x0 - mu) * inv * g0 + c0, 0.f);
            float y1 = fmaxf((x1 - mu) * inv * g1 + c1, 0.f);
            float d = y0 * w0 + y1 * w1;
#pragma unroll
            for (int off = 16; off > 0; off >>= 1) d += __shfl_xor(d, off, 32);
            if (lane == 0 && row < nrows) preds[rowIdx[row]] = d + bo;
        }
    }
}

// ---------------------------------------------------------------------------
extern "C" void kernel_launch(void* const* d_in, const int* in_sizes, int n_in,
                              void* d_out, int out_size, void* d_ws, size_t ws_size,
                              hipStream_t stream) {
    const float* x      = (const float*)d_in[0];
    const int*   labels = (const int*)d_in[1];
    const float *W0 = (const float*)d_in[2],  *b0 = (const float*)d_in[3],
                *g0 = (const float*)d_in[4],  *be0 = (const float*)d_in[5],
                *m0 = (const float*)d_in[6],  *v0 = (const float*)d_in[7];
    const float *W1 = (const float*)d_in[8],  *b1 = (const float*)d_in[9],
                *g1 = (const float*)d_in[10], *be1 = (const float*)d_in[11],
                *m1 = (const float*)d_in[12], *v1 = (const float*)d_in[13];
    const float *W2 = (const float*)d_in[14], *b2 = (const float*)d_in[15],
                *g2 = (const float*)d_in[16], *be2 = (const float*)d_in[17],
                *m2 = (const float*)d_in[18], *v2 = (const float*)d_in[19];
    const float *HW1 = (const float*)d_in[20], *Hb1 = (const float*)d_in[21],
                *LG1 = (const float*)d_in[22], *LB1 = (const float*)d_in[23];
    const float *HW2 = (const float*)d_in[24], *Hb2 = (const float*)d_in[25],
                *LG2 = (const float*)d_in[26], *LB2 = (const float*)d_in[27];
    const float *HWo = (const float*)d_in[28], *Hbo = (const float*)d_in[29];
    float* preds = (float*)d_out;

    char* ws = (char*)d_ws;
    size_t off = 0;
    auto carve = [&](size_t bytes) {
        char* p = ws + off;
        off = (off + bytes + 255) & ~(size_t)255;
        return p;
    };
    __bf16* h0hi = (__bf16*)carve((size_t)BCELLS * 512 * 2);
    __bf16* h0lo = (__bf16*)carve((size_t)BCELLS * 512 * 2);
    __bf16* h1hi = (__bf16*)carve((size_t)BCELLS * 256 * 2);
    __bf16* h1lo = (__bf16*)carve((size_t)BCELLS * 256 * 2);
    __bf16* fHi  = h0hi;  // h0 planes dead once h1 exists -> reuse for feats
    __bf16* fLo  = h0lo;
    int* counts  = (int*)carve(32 * sizeof(int));
    int* lists   = (int*)carve((size_t)NEXP * BCELLS * sizeof(int));
    __bf16* W0p = (__bf16*)carve((size_t)63 * 4 * 2 * 128 * SWB * 2);
    __bf16* W1p = (__bf16*)carve((size_t)16 * 2 * 2 * 128 * SWB * 2);
    __bf16* W2p = (__bf16*)carve((size_t)8  * 2 * 2 * 128 * SWB * 2);
    __bf16* H1p = (__bf16*)carve((size_t)NEXP * 8 * 2 * 128 * SWB * 2);
    __bf16* H2p = (__bf16*)carve((size_t)NEXP * 4 * 2 * 64  * SWB * 2);

    (void)hipMemsetAsync(counts, 0, 32 * sizeof(int), stream);
    route_kernel<<<(BCELLS + 255) / 256, 256, 0, stream>>>(labels, counts, lists);

    pack_w_kernel<<<256, 256, 0, stream>>>(W0,  W0p, 1985, 512, 128, 1);
    pack_w_kernel<<<256, 256, 0, stream>>>(W1,  W1p, 512,  256, 128, 1);
    pack_w_kernel<<<256, 256, 0, stream>>>(W2,  W2p, 256,  256, 128, 1);
    pack_w_kernel<<<256, 256, 0, stream>>>(HW1, H1p, 256,  128, 128, NEXP);
    pack_w_kernel<<<256, 256, 0, stream>>>(HW2, H2p, 128,  64,  64,  NEXP);

    gemm_f32in_kernel<1985, 512><<<dim3(4, BCELLS / 128), 256, 0, stream>>>(
        x, W0p, b0, g0, be0, m0, v0, h0hi, h0lo);
    gemm_bf_kernel<512, 256><<<dim3(2, BCELLS / 128), 256, 0, stream>>>(
        h0hi, h0lo, W1p, b1, g1, be1, m1, v1, h1hi, h1lo);
    gemm_bf_kernel<256, 256><<<dim3(2, BCELLS / 128), 256, 0, stream>>>(
        h1hi, h1lo, W2p, b2, g2, be2, m2, v2, fHi, fLo);

    head_kernel<<<dim3(BCELLS / 64, NEXP), 256, 0, stream>>>(
        fHi, fLo, counts, lists, H1p, Hb1, LG1, LB1, H2p, Hb2, LG2, LB2,
        HWo, Hbo, preds);
}
